// GenerativeModel_83288005804620
// MI455X (gfx1250) — matrix operand order repaired and checked by reading, not compiled, not run on hardware
//
#include <hip/hip_runtime.h>

// ---------------------------------------------------------------------------
// Problem constants (from reference)
// ---------------------------------------------------------------------------
#define N_NODES  12800
#define N_EDGES  204800
#define L_SEQ    24
#define H_DIM    128
#define B_BATCH  64
#define R_REL    8
#define VN_TYPES 100
#define VA_ACT   20
#define NODES_PER_B (N_NODES / B_BATCH)   // 200
#define K_RGCN   (R_REL * H_DIM + H_DIM)  // 1152 (relation 1024 + root 128)

typedef _Float16 f16;
typedef __attribute__((ext_vector_type(16))) _Float16 v16h;
typedef __attribute__((ext_vector_type(8)))  _Float16 v8h;
typedef __attribute__((ext_vector_type(8)))  float    v8f;

// sched_group_barrier masks
#define SGB_WMMA      0x008
#define SGB_VMEM_READ 0x020

// ---------------------------------------------------------------------------
// Elementwise helpers
// ---------------------------------------------------------------------------
__global__ void zero_kernel(float* __restrict__ p, int n) {
    int i = blockIdx.x * blockDim.x + threadIdx.x;
    if (i < n) p[i] = 0.0f;
}

// x[n,h] = emb_nodes_w[nodeTypes[n],h]  (f32 + f16 copy)
__global__ void embed_kernel(const int* __restrict__ nodeTypes,
                             const float* __restrict__ emb_w,
                             float* __restrict__ x, f16* __restrict__ xh) {
    int i = blockIdx.x * blockDim.x + threadIdx.x;
    if (i >= N_NODES * H_DIM) return;
    int n = i >> 7, h = i & 127;
    float v = emb_w[nodeTypes[n] * H_DIM + h];
    x[i] = v;
    xh[i] = (f16)v;
}

// ---------------------------------------------------------------------------
// Weight packing directly into WMMA B-fragment layout.
// Fragment layout: BF[((td*KT32 + kb32)*32 + lane)*16 + i]
//   holds logical B[k, d] with k = kb32*32 + (lane>>4)*16 + i,
//                             d = td*16 + (lane&15)
// so each lane reads its 16-half fragment as one contiguous 32B chunk.
// ---------------------------------------------------------------------------
// RGCN: B[k,d] = (k<1024) ? rgcn_W.flat[k*128+d] : rgcn_root[(k-1024)*128+d]
__global__ void pack_wcat_frag(const float* __restrict__ rgcn_W,
                               const float* __restrict__ rgcn_root,
                               f16* __restrict__ BF) {
    const int KT32 = K_RGCN / 32;                 // 36
    int idx = blockIdx.x * blockDim.x + threadIdx.x;
    if (idx >= K_RGCN * H_DIM) return;
    int i    = idx & 15;
    int lane = (idx >> 4) & 31;
    int rem  = idx >> 9;                          // td*KT32 + kb32
    int td   = rem / KT32;
    int kb32 = rem - td * KT32;
    int k = kb32 * 32 + (lane >> 4) * 16 + i;
    int d = td * 16 + (lane & 15);
    float v = (k < R_REL * H_DIM) ? rgcn_W[k * H_DIM + d]
                                  : rgcn_root[(k - R_REL * H_DIM) * H_DIM + d];
    BF[idx] = (f16)v;
}

// part2: B[k,d] = linN_w[d, H + k]  (transpose of second half)
__global__ void pack_linN2_frag(const float* __restrict__ linN_w,
                                f16* __restrict__ BF) {
    const int KT32 = H_DIM / 32;                  // 4
    int idx = blockIdx.x * blockDim.x + threadIdx.x;
    if (idx >= H_DIM * H_DIM) return;
    int i    = idx & 15;
    int lane = (idx >> 4) & 31;
    int rem  = idx >> 9;
    int td   = rem / KT32;
    int kb32 = rem - td * KT32;
    int k = kb32 * 32 + (lane >> 4) * 16 + i;
    int d = td * 16 + (lane & 15);
    BF[idx] = (f16)linN_w[d * (2 * H_DIM) + H_DIM + k];
}

// ---------------------------------------------------------------------------
// RGCN edge scatter: msg[(dst*R+et), h] += x[src, h];  cnt[(dst*R+et)] += 1
// ---------------------------------------------------------------------------
__global__ void scatter_kernel(const int* __restrict__ src,
                               const int* __restrict__ dst,
                               const int* __restrict__ et,
                               const float* __restrict__ x,
                               float* __restrict__ msg,
                               float* __restrict__ cnt) {
    int e = blockIdx.x;
    int h = threadIdx.x;
    int s = src[e], d = dst[e], r = et[e];
    size_t seg = (size_t)d * R_REL + r;
    atomicAdd(&msg[seg * H_DIM + h], x[(size_t)s * H_DIM + h]);
    if (h == 0) atomicAdd(&cnt[seg], 1.0f);
}

// meanh = f16( msg / max(cnt,1) )
__global__ void mean_cvt_kernel(const float* __restrict__ msg,
                                const float* __restrict__ cnt,
                                f16* __restrict__ meanh) {
    int i = blockIdx.x * blockDim.x + threadIdx.x;
    if (i >= N_NODES * R_REL * H_DIM) return;
    float c = cnt[i >> 7];
    if (c < 1.0f) c = 1.0f;
    meanh[i] = (f16)(msg[i] / c);
}

// ---------------------------------------------------------------------------
// WMMA GEMM: C[M,128] = act( concat(A1[M,K1], A2[M,K2]) @ B[K1+K2,128] + bias )
// One wave owns a 16-row x 64-col strip (4 accumulator tiles -> 2 waves per
// band, 2x wave parallelism, ~96 live VGPRs). Per K-step the schedule is
// pinned with sched_group_barrier: 10 VMEM reads (8 B-fragment b128 + 2
// A-prefetch b128) issue FIRST, then the 4 WMMAs -- so load waits stagger
// (<=8,<=6,<=4,<=2) instead of draining to 0, and the next-step A fragment's
// latency hides under the WMMA burst. 4 independent accumulators also cover
// the WMMA->WMMA RAW hazard.
// ---------------------------------------------------------------------------
template<int K1, int K2, bool BIAS, bool RELU, bool W16>
__global__ void __launch_bounds__(256)
wmma_gemm_kernel(const f16* __restrict__ A1,
                 const f16* __restrict__ A2,
                 const f16* __restrict__ BF,
                 const float* __restrict__ bias,
                 float* __restrict__ Cout,
                 f16* __restrict__ Cout16,
                 int M) {
    constexpr int Ktot = K1 + K2;
    constexpr int KT32 = Ktot / 32;
    constexpr int TD   = 4;                       // tiles per wave (64 cols)
    int wave = (blockIdx.x * blockDim.x + threadIdx.x) >> 5;
    int lane = threadIdx.x & 31;
    int tm  = wave >> 1;                          // 16-row band
    int tdg = wave & 1;                           // column-half (0..1)
    int m0 = tm << 4;
    if (m0 >= M) return;                 // wave-uniform exit (EXEC stays full)
    int hi  = lane >> 4;                 // 0 or 1
    int row = m0 + (lane & 15);

    // per-lane A chunk addresses for K-step kb: two contiguous 8-half runs at
    // k = kb + hi*16 and k = kb + hi*16 + 8 (contiguous order, matches packed B)
    auto aptr = [&](int k) -> const f16* {
        return (k < K1) ? (A1 + (size_t)row * K1 + k)
                        : (A2 + (size_t)row * K2 + (k - K1));
    };

    v8f c[TD];
#pragma unroll
    for (int t = 0; t < TD; ++t) c[t] = (v8f){0.f,0.f,0.f,0.f,0.f,0.f,0.f,0.f};

    // base of this wave's column group inside the fragment-packed B
    const f16* BFg = BF + (size_t)(tdg * TD) * KT32 * 32 * 16;

    // prologue: A fragment for step 0
    v8h lo = *(const v8h*)aptr(hi * 16);
    v8h hh = *(const v8h*)aptr(8 + hi * 16);

    for (int kb32 = 0; kb32 < KT32; ++kb32) {
        v16h a;
#pragma unroll
        for (int i = 0; i < 8; ++i) { a[i] = lo[i]; a[i + 8] = hh[i]; }

        // B-fragment loads for this K-step (8 x b128)
        v16h bf[TD];
#pragma unroll
        for (int t = 0; t < TD; ++t)
            bf[t] = *(const v16h*)(BFg +
                      (((size_t)t * KT32 + kb32) * 32 + lane) * 16);

        // unconditional A prefetch (clamped on last step -> exact group size)
        int kbn = (kb32 + 1 < KT32) ? (kb32 + 1) * 32 : 0;
        lo = *(const v8h*)aptr(kbn + hi * 16);
        hh = *(const v8h*)aptr(kbn + 8 + hi * 16);

#pragma unroll
        for (int t = 0; t < TD; ++t)
            c[t] = __builtin_amdgcn_wmma_f32_16x16x32_f16(
                       false, a, false, bf[t], (short)0, c[t], false, false);

        // pin schedule: all 10 VMEM reads first, then the 4 WMMAs
        __builtin_amdgcn_sched_group_barrier(SGB_VMEM_READ, 10, 0);
        __builtin_amdgcn_sched_group_barrier(SGB_WMMA,       4, 0);
    }

#pragma unroll
    for (int t = 0; t < TD; ++t) {
        int col = (tdg * TD + t) * 16 + (lane & 15);
        float bv = BIAS ? bias[col] : 0.0f;
#pragma unroll
        for (int j = 0; j < 8; ++j) {
            int r = m0 + j + hi * 8;
            float v = c[t][j] + bv;
            if (RELU) v = fmaxf(v, 0.0f);
            Cout[(size_t)r * H_DIM + col] = v;
            if (W16) Cout16[(size_t)r * H_DIM + col] = (f16)v;
        }
    }
}

// ---------------------------------------------------------------------------
// h_G[b,h] = mean over 200 contiguous nodes
// ---------------------------------------------------------------------------
__global__ void hg_kernel(const float* __restrict__ x, float* __restrict__ hG) {
    int b = blockIdx.x, h = threadIdx.x;
    float s = 0.0f;
    for (int i = 0; i < NODES_PER_B; ++i)
        s += x[((size_t)b * NODES_PER_B + i) * H_DIM + h];
    hG[b * H_DIM + h] = s * (1.0f / NODES_PER_B);
}

// ---------------------------------------------------------------------------
// action + final heads (tiny): block per batch, 128 threads
// ---------------------------------------------------------------------------
__global__ void heads_kernel(const float* __restrict__ hG,
                             const float* __restrict__ linA_w, const float* __restrict__ linA_b,
                             const float* __restrict__ linAf_w, const float* __restrict__ linAf_b,
                             const float* __restrict__ fin_w, const float* __restrict__ fin_b,
                             const float* __restrict__ finf_w, const float* __restrict__ finf_b,
                             float* __restrict__ out_action, float* __restrict__ out_final) {
    int b = blockIdx.x, j = threadIdx.x;
    __shared__ float hg[H_DIM];
    __shared__ float hid[H_DIM];
    hg[j] = hG[b * H_DIM + j];
    __syncthreads();

    float acc = linA_b[j];
    for (int h = 0; h < H_DIM; ++h) acc += hg[h] * linA_w[j * H_DIM + h];
    hid[j] = fmaxf(acc, 0.0f);
    __syncthreads();
    if (j < VA_ACT) {
        float a = linAf_b[j];
        for (int h = 0; h < H_DIM; ++h) a += hid[h] * linAf_w[j * H_DIM + h];
        out_action[b * VA_ACT + j] = a;
    }
    __syncthreads();

    float acc2 = fin_b[j];
    for (int h = 0; h < H_DIM; ++h) acc2 += hg[h] * fin_w[j * H_DIM + h];
    hid[j] = fmaxf(acc2, 0.0f);
    __syncthreads();
    if (j == 0) {
        float a = finf_b[0];
        for (int h = 0; h < H_DIM; ++h) a += hid[h] * finf_w[h];
        out_final[b] = 1.0f / (1.0f + __expf(-a));
    }
}

// ---------------------------------------------------------------------------
// agg[b,l,h] = sum over 200 nodes of seqin[n,l] * nodeEmb[n,h]
// ---------------------------------------------------------------------------
__global__ void agg_kernel(const float* __restrict__ seqin,
                           const float* __restrict__ x,
                           float* __restrict__ agg) {
    int blk = blockIdx.x;
    int b = blk / L_SEQ, l = blk - b * L_SEQ;
    int h = threadIdx.x;
    float s = 0.0f;
    for (int i = 0; i < NODES_PER_B; ++i) {
        int n = b * NODES_PER_B + i;
        s += seqin[n * L_SEQ + l] * x[(size_t)n * H_DIM + h];
    }
    agg[(size_t)blk * H_DIM + h] = s;
}

// ---------------------------------------------------------------------------
// GRU (torch gate order r,z,n). Block per batch, 128 threads, sequential in t.
// ---------------------------------------------------------------------------
__global__ void gru_kernel(const float* __restrict__ hG,
                           const float* __restrict__ agg,
                           const float* __restrict__ emb_actions_w,
                           const int* __restrict__ action_input,
                           const float* __restrict__ Wih, const float* __restrict__ Whh,
                           const float* __restrict__ bih, const float* __restrict__ bhh,
                           float* __restrict__ seqout) {
    int b = blockIdx.x, j = threadIdx.x;
    __shared__ float h[H_DIM];
    __shared__ float xt[H_DIM];
    h[j] = hG[b * H_DIM + j];
    __syncthreads();
    for (int t = 0; t < L_SEQ; ++t) {
        xt[j] = (t == 0) ? emb_actions_w[action_input[b] * H_DIM + j]
                         : agg[((size_t)b * L_SEQ + (t - 1)) * H_DIM + j];
        __syncthreads();
        float ir = bih[j], iz = bih[H_DIM + j], in_ = bih[2 * H_DIM + j];
        float hr = bhh[j], hz = bhh[H_DIM + j], hn = bhh[2 * H_DIM + j];
        for (int k = 0; k < H_DIM; ++k) {
            float xv = xt[k], hv = h[k];
            ir  += xv * Wih[j * H_DIM + k];
            iz  += xv * Wih[(H_DIM + j) * H_DIM + k];
            in_ += xv * Wih[(2 * H_DIM + j) * H_DIM + k];
            hr  += hv * Whh[j * H_DIM + k];
            hz  += hv * Whh[(H_DIM + j) * H_DIM + k];
            hn  += hv * Whh[(2 * H_DIM + j) * H_DIM + k];
        }
        float r = 1.0f / (1.0f + __expf(-(ir + hr)));
        float z = 1.0f / (1.0f + __expf(-(iz + hz)));
        float nn = tanhf(in_ + r * hn);
        float hnew = (1.0f - z) * nn + z * h[j];
        __syncthreads();
        h[j] = hnew;
        seqout[((size_t)b * L_SEQ + t) * H_DIM + j] = hnew;
        __syncthreads();
    }
}

// part1[row,o] = dot(seq[row,:], linN_w[o, 0:H]) ; row = b*L + l
__global__ void part1_kernel(const float* __restrict__ seq,
                             const float* __restrict__ linN_w,
                             float* __restrict__ part1) {
    int row = blockIdx.x, o = threadIdx.x;
    __shared__ float sv[H_DIM];
    sv[o] = seq[(size_t)row * H_DIM + o];
    __syncthreads();
    float s = 0.0f;
    for (int k = 0; k < H_DIM; ++k) s += sv[k] * linN_w[o * (2 * H_DIM) + k];
    part1[(size_t)row * H_DIM + o] = s;
}

// logits[n,l] = dot(relu(part1[bs[n],l,:] + part2[n,:] + linN_b), linNf_w) + linNf_b
__global__ void logits_kernel(const float* __restrict__ part1,
                              const float* __restrict__ part2,
                              const float* __restrict__ linN_b,
                              const float* __restrict__ linNf_w,
                              const float* __restrict__ linNf_b,
                              float* __restrict__ logits) {
    int n = blockIdx.x, h = threadIdx.x;
    int b = n / NODES_PER_B;
    __shared__ float red[H_DIM];
    float p2 = part2[(size_t)n * H_DIM + h] + linN_b[h];
    float w = linNf_w[h];
    for (int l = 0; l < L_SEQ; ++l) {
        float v = part1[((size_t)b * L_SEQ + l) * H_DIM + h] + p2;
        red[h] = fmaxf(v, 0.0f) * w;
        __syncthreads();
        for (int s = 64; s > 0; s >>= 1) {
            if (h < s) red[h] += red[h + s];
            __syncthreads();
        }
        if (h == 0) logits[n * L_SEQ + l] = red[0] + linNf_b[0];
        __syncthreads();
    }
}

// per-(batch,l) softmax over the 200 contiguous nodes
__global__ void segsoftmax_kernel(const float* __restrict__ logits,
                                  float* __restrict__ out_nodes) {
    int blk = blockIdx.x;
    int b = blk / L_SEQ, l = blk - b * L_SEQ;
    int t = threadIdx.x;                 // 256 threads
    __shared__ float red[256];
    float v = -1e30f;
    if (t < NODES_PER_B) v = logits[(b * NODES_PER_B + t) * L_SEQ + l];
    red[t] = v;
    __syncthreads();
    for (int s = 128; s > 0; s >>= 1) {
        if (t < s) red[t] = fmaxf(red[t], red[t + s]);
        __syncthreads();
    }
    float m = red[0];
    __syncthreads();
    float e = (t < NODES_PER_B) ? __expf(v - m) : 0.0f;
    red[t] = e;
    __syncthreads();
    for (int s = 128; s > 0; s >>= 1) {
        if (t < s) red[t] += red[t + s];
        __syncthreads();
    }
    float ssum = red[0];
    if (t < NODES_PER_B)
        out_nodes[(b * NODES_PER_B + t) * L_SEQ + l] = e / ssum;
}

// ---------------------------------------------------------------------------
// Host launch
// ---------------------------------------------------------------------------
extern "C" void kernel_launch(void* const* d_in, const int* in_sizes, int n_in,
                              void* d_out, int out_size, void* d_ws, size_t ws_size,
                              hipStream_t stream) {
    const int*   nodeTypes   = (const int*)d_in[0];
    const int*   edge_index  = (const int*)d_in[1];   // [2,E]
    const int*   edge_attr   = (const int*)d_in[2];
    const float* seqin       = (const float*)d_in[4]; // [N,L]
    const int*   action_in   = (const int*)d_in[7];
    const float* emb_nodes_w = (const float*)d_in[8];
    const float* emb_act_w   = (const float*)d_in[9];
    const float* rgcn_W      = (const float*)d_in[10];
    const float* rgcn_root   = (const float*)d_in[11];
    const float* rgcn_b      = (const float*)d_in[12];
    const float* gru_Wih     = (const float*)d_in[13];
    const float* gru_Whh     = (const float*)d_in[14];
    const float* gru_bih     = (const float*)d_in[15];
    const float* gru_bhh     = (const float*)d_in[16];
    const float* linA_w      = (const float*)d_in[17];
    const float* linA_b      = (const float*)d_in[18];
    const float* linAf_w     = (const float*)d_in[19];
    const float* linAf_b     = (const float*)d_in[20];
    const float* linN_w      = (const float*)d_in[21];
    const float* linN_b      = (const float*)d_in[22];
    const float* linNf_w     = (const float*)d_in[23];
    const float* linNf_b     = (const float*)d_in[24];
    const float* fin_w       = (const float*)d_in[25];
    const float* fin_b       = (const float*)d_in[26];
    const float* finf_w      = (const float*)d_in[27];
    const float* finf_b      = (const float*)d_in[28];

    const int* esrc = edge_index;
    const int* edst = edge_index + N_EDGES;

    // output layout: action [B*VA] | nodes_final [N*L] | final [B]
    float* out_action = (float*)d_out;
    float* out_nodes  = out_action + B_BATCH * VA_ACT;
    float* out_final  = out_nodes + (size_t)N_NODES * L_SEQ;

    // workspace carve-out
    char* ws = (char*)d_ws;
    size_t off = 0;
    auto carve = [&](size_t bytes) -> void* {
        void* p = ws + off;
        off = (off + bytes + 255) & ~(size_t)255;
        return p;
    };
    float* msg    = (float*)carve((size_t)N_NODES * R_REL * H_DIM * 4);
    float* cnt    = (float*)carve((size_t)N_NODES * R_REL * 4);
    float* xA     = (float*)carve((size_t)N_NODES * H_DIM * 4);
    float* xB     = (float*)carve((size_t)N_NODES * H_DIM * 4);
    f16*   xhA    = (f16*)  carve((size_t)N_NODES * H_DIM * 2);
    f16*   xhB    = (f16*)  carve((size_t)N_NODES * H_DIM * 2);
    f16*   meanh  = (f16*)  carve((size_t)N_NODES * R_REL * H_DIM * 2);
    f16*   BFrgcn = (f16*)  carve((size_t)K_RGCN * H_DIM * 2);
    f16*   BFlinN = (f16*)  carve((size_t)H_DIM * H_DIM * 2);
    float* hG     = (float*)carve((size_t)B_BATCH * H_DIM * 4);
    float* agg    = (float*)carve((size_t)B_BATCH * L_SEQ * H_DIM * 4);
    float* seqout = (float*)carve((size_t)B_BATCH * L_SEQ * H_DIM * 4);
    float* part1  = (float*)carve((size_t)B_BATCH * L_SEQ * H_DIM * 4);
    float* part2  = (float*)carve((size_t)N_NODES * H_DIM * 4);
    float* logits = (float*)carve((size_t)N_NODES * L_SEQ * 4);

    const int NH  = N_NODES * H_DIM;          // 1,638,400
    const int NRH = N_NODES * R_REL * H_DIM;  // 13,107,200

    // weight packing into WMMA fragment layout (f16)
    pack_wcat_frag<<<(K_RGCN * H_DIM + 255) / 256, 256, 0, stream>>>(rgcn_W, rgcn_root, BFrgcn);
    pack_linN2_frag<<<(H_DIM * H_DIM + 255) / 256, 256, 0, stream>>>(linN_w, BFlinN);

    // node-type embedding
    embed_kernel<<<(NH + 255) / 256, 256, 0, stream>>>(nodeTypes, emb_nodes_w, xA, xhA);

    // two shared-weight RGCN layers + ReLU (ping-pong xA<->xB)
    float* xin = xA;  f16* xhin = xhA;
    float* xout = xB; f16* xhout = xhB;
    const int gemmWaves  = (N_NODES / 16) * 2;    // 1600 (wave = 16x64 strip)
    const int gemmBlocks = (gemmWaves + 7) / 8;   // 256 thr = 8 waves
    for (int layer = 0; layer < 2; ++layer) {
        zero_kernel<<<(NRH + 255) / 256, 256, 0, stream>>>(msg, NRH);
        zero_kernel<<<(N_NODES * R_REL + 255) / 256, 256, 0, stream>>>(cnt, N_NODES * R_REL);
        scatter_kernel<<<N_EDGES, H_DIM, 0, stream>>>(esrc, edst, edge_attr, xin, msg, cnt);
        mean_cvt_kernel<<<(NRH + 255) / 256, 256, 0, stream>>>(msg, cnt, meanh);
        wmma_gemm_kernel<R_REL * H_DIM, H_DIM, true, true, true>
            <<<gemmBlocks, 256, 0, stream>>>(
                meanh, xhin, BFrgcn, rgcn_b, xout, xhout, N_NODES);
        // swap
        float* tf = xin;  xin = xout;  xout = tf;
        f16*   th = xhin; xhin = xhout; xhout = th;
    }
    // nodeEmb now in xin (== xA), f16 copy in xhin (== xhA)
    const float* nodeEmb  = xin;
    const f16*   nodeEmbH = xhin;

    // graph readout + heads
    hg_kernel<<<B_BATCH, H_DIM, 0, stream>>>(nodeEmb, hG);
    heads_kernel<<<B_BATCH, H_DIM, 0, stream>>>(hG, linA_w, linA_b, linAf_w, linAf_b,
                                                fin_w, fin_b, finf_w, finf_b,
                                                out_action, out_final);

    // per-timestep weighted node aggregation, then GRU
    agg_kernel<<<B_BATCH * L_SEQ, H_DIM, 0, stream>>>(seqin, nodeEmb, agg);
    gru_kernel<<<B_BATCH, H_DIM, 0, stream>>>(hG, agg, emb_act_w, action_in,
                                              gru_Wih, gru_Whh, gru_bih, gru_bhh, seqout);

    // node scoring
    part1_kernel<<<B_BATCH * L_SEQ, H_DIM, 0, stream>>>(seqout, linN_w, part1);
    wmma_gemm_kernel<H_DIM, 0, false, false, false>
        <<<gemmBlocks, 256, 0, stream>>>(
            nodeEmbH, (const f16*)nullptr, BFlinN, (const float*)nullptr,
            part2, (f16*)nullptr, N_NODES);
    logits_kernel<<<N_NODES, H_DIM, 0, stream>>>(part1, part2, linN_b,
                                                 linNf_w, linNf_b, logits);
    segsoftmax_kernel<<<B_BATCH * L_SEQ, 256, 0, stream>>>(logits, out_nodes);
}
